// TripletLoss_21337397527200
// MI455X (gfx1250) — compile-verified, matrix operand order checked by reading, not run on hardware
//
#include <hip/hip_runtime.h>
#include <hip/hip_bf16.h>

typedef __attribute__((ext_vector_type(16))) __bf16 v16bf;
typedef __attribute__((ext_vector_type(8)))  __bf16 v8bf;
typedef __attribute__((ext_vector_type(8)))  float  v8f;
typedef __attribute__((ext_vector_type(4)))  float  v4f;
typedef __attribute__((ext_vector_type(8)))  unsigned short v8us;
typedef __attribute__((ext_vector_type(4)))  unsigned int v4u;
typedef __attribute__((ext_vector_type(8)))  int v8i;
typedef __attribute__((ext_vector_type(4)))  int v4i;

#define N_ROWS 4096
#define DIM    2048
#define KID    4
#define NPAIR  6                         // K*(K-1)/2
#define NP_TOT (N_ROWS / KID * NPAIR)    // 6144
#define MARGIN 0.3f

// GEMM tiling
#define BM 128
#define BN 64
#define BK 32
#define KSTEPS (DIM / BK)
#define LDK 40   // padded k-stride (bf16 elems): 80B rows -> conflict-free b128 LDS reads

// workspace layout (bytes)
#define WS_SQ      0
#define WS_NEGMIN  (N_ROWS * 4)
#define WS_AP      (2 * N_ROWS * 4)
#define WS_HI      65536
#define WS_LO      (WS_HI + (size_t)N_ROWS * DIM * 2)
#define WS_NEED    (WS_LO + (size_t)N_ROWS * DIM * 2)

#if defined(__gfx1250__) && __has_builtin(__builtin_amdgcn_tensor_load_to_lds) && \
    __has_builtin(__builtin_amdgcn_s_wait_tensorcnt)
#define HAVE_TDM 1
#else
#define HAVE_TDM 0
#endif

__device__ __forceinline__ unsigned short f2bf(float f) {
    unsigned u = __float_as_uint(f);
    unsigned r = u + 0x7FFFu + ((u >> 16) & 1u);   // round-to-nearest-even
    return (unsigned short)(r >> 16);
}
__device__ __forceinline__ float bf2f(unsigned short h) {
    return __uint_as_float(((unsigned)h) << 16);
}

__device__ __forceinline__ v16bf frag16(const unsigned short* base, int kb) {
    v8bf h0 = *(const v8bf*)(base + kb);
    v8bf h1 = *(const v8bf*)(base + kb + 16);
    return __builtin_shufflevector(h0, h1, 0,1,2,3,4,5,6,7,8,9,10,11,12,13,14,15);
}

// Shared WMMA micro-kernel: 12 x v_wmma_f32_16x16x32_bf16 per wave per k-step
__device__ __forceinline__ void wave_mma_step(
        const unsigned short* As_hi, const unsigned short* As_lo,
        const unsigned short* Bs_hi, const unsigned short* Bs_lo,
        int wm, int wn, int lrow, int kb, v8f acc[2][2]) {
    v16bf a_hi[2], a_lo[2], b_hi[2], b_lo[2];
    #pragma unroll
    for (int g = 0; g < 2; ++g) {
        int ar = wm + g * 16 + lrow;
        int br = wn + g * 16 + lrow;
        a_hi[g] = frag16(&As_hi[ar * LDK], kb);
        a_lo[g] = frag16(&As_lo[ar * LDK], kb);
        b_hi[g] = frag16(&Bs_hi[br * LDK], kb);
        b_lo[g] = frag16(&Bs_lo[br * LDK], kb);
    }
    #pragma unroll
    for (int ag = 0; ag < 2; ++ag)
        #pragma unroll
        for (int bg = 0; bg < 2; ++bg) {
            acc[ag][bg] = __builtin_amdgcn_wmma_f32_16x16x32_bf16(
                false, a_hi[ag], false, b_hi[bg], (short)0, acc[ag][bg], false, false);
            acc[ag][bg] = __builtin_amdgcn_wmma_f32_16x16x32_bf16(
                false, a_hi[ag], false, b_lo[bg], (short)0, acc[ag][bg], false, false);
            acc[ag][bg] = __builtin_amdgcn_wmma_f32_16x16x32_bf16(
                false, a_lo[ag], false, b_hi[bg], (short)0, acc[ag][bg], false, false);
        }
}

// Shared epilogue: dist = sqrt(sq_m + sq_n - 2G), mask same-target, lane-min, atomicMin
__device__ __forceinline__ void wave_epilogue(
        const v8f acc[2][2], const float* sq, const int* tgt, unsigned* negmin,
        int bm, int bn, int wm, int wn, int lane, int lrow) {
    #pragma unroll
    for (int ag = 0; ag < 2; ++ag) {
        #pragma unroll
        for (int bg = 0; bg < 2; ++bg) {
            int   n     = bn + wn + bg * 16 + lrow;
            float sqn   = sq[n];
            int   tn    = tgt[n];
            int   mbase = bm + wm + ag * 16 + ((lane < 16) ? 0 : 8);
            #pragma unroll
            for (int v = 0; v < 8; ++v) {
                int m = mbase + v;
                float d2 = sq[m] + sqn - 2.0f * acc[ag][bg][v];
                float d  = sqrtf(fmaxf(d2, 1e-12f));
                float dm = (tgt[m] == tn) ? __builtin_inff() : d;
                #pragma unroll
                for (int off = 8; off >= 1; off >>= 1)
                    dm = fminf(dm, __shfl_xor(dm, off, 32));
                if (lrow == 0)
                    atomicMin(&negmin[m], __float_as_uint(dm));
            }
        }
    }
}

#if HAVE_TDM
// Issue a 2-D TDM tile load: rows x 32 bf16 elems, row stride DIM, with LDS
// padding 16B per 64B row (reproduces the LDK=40 conflict-free layout).
__device__ __forceinline__ void tdm_load_2d(unsigned lds_addr,
                                            const unsigned short* gptr,
                                            int tile_rows) {
    unsigned long long ga = (unsigned long long)(uintptr_t)gptr;
    v4u g0;
    g0[0] = 1u;                                            // count=1, gather off
    g0[1] = lds_addr;                                      // LDS dest (bytes)
    g0[2] = (unsigned)(ga & 0xFFFFFFFFull);                // global_addr[31:0]
    g0[3] = (unsigned)((ga >> 32) & 0x01FFFFFFull)         // global_addr[56:32]
            | (2u << 30);                                  // type=2 (image)
    v8i g1;
    g1[0] = (int)(0x00010000u                              // data_size=1 -> 2B
            | (1u << 20)                                   // pad_enable
            | (3u << 22)                                   // pad_interval: 16 DW
            | (3u << 25));                                 // pad_amount: 4 DW (16B)
    g1[1] = (int)0xFFFF0000u;                              // tensor_dim0 lo16 (huge)
    g1[2] = (int)(0x7FFFu | 0xFFFF0000u);                  // dim0 hi | dim1 lo16
    g1[3] = (int)(0x7FFFu | (32u << 16));                  // dim1 hi | tile_dim0=32
    g1[4] = tile_rows;                                     // tile_dim1 | tile_dim2=0
    g1[5] = DIM;                                           // tensor_dim0_stride lo32
    g1[6] = 0;
    g1[7] = 0;
    v4i z = {0, 0, 0, 0};
#if __clang_major__ >= 23
    v8i z8 = {0, 0, 0, 0, 0, 0, 0, 0};
    __builtin_amdgcn_tensor_load_to_lds(g0, g1, z, z, z8, 0);
#else
    __builtin_amdgcn_tensor_load_to_lds(g0, g1, z, z, 0);
#endif
}
#endif

// ---------------------------------------------------------------------------
// Kernel 0: split fp32 X into bf16 hi + bf16 lo planes (one pass, HBM-bound)
// ---------------------------------------------------------------------------
__global__ __launch_bounds__(256)
void xsplit(const float* __restrict__ x, unsigned short* __restrict__ hi,
            unsigned short* __restrict__ lo) {
    size_t base = ((size_t)blockIdx.x * 256 + threadIdx.x) * 8;
    v4f v0 = *(const v4f*)(x + base);
    v4f v1 = *(const v4f*)(x + base + 4);
    v8us h, l;
    #pragma unroll
    for (int e = 0; e < 4; ++e) {
        unsigned short hh0 = f2bf(v0[e]);
        unsigned short hh1 = f2bf(v1[e]);
        h[e]     = hh0;
        h[e + 4] = hh1;
        l[e]     = f2bf(v0[e] - bf2f(hh0));
        l[e + 4] = f2bf(v1[e] - bf2f(hh1));
    }
    *(v8us*)(hi + base) = h;
    *(v8us*)(lo + base) = l;
}

// ---------------------------------------------------------------------------
// Kernel 1: per-row squared norms + init neg-min to +inf bits
// ---------------------------------------------------------------------------
__global__ __launch_bounds__(256)
void rownorm_init(const float* __restrict__ x, float* __restrict__ sq,
                  unsigned* __restrict__ negmin) {
    const int row = blockIdx.x;
    const float* xr = x + (size_t)row * DIM;
    float s = 0.f;
    for (int k = threadIdx.x * 4; k < DIM; k += 256 * 4) {
        v4f v = *(const v4f*)(xr + k);
        s += v.x * v.x + v.y * v.y + v.z * v.z + v.w * v.w;
    }
    __shared__ float red[256];
    red[threadIdx.x] = s;
    __syncthreads();
    for (int off = 128; off > 0; off >>= 1) {
        if (threadIdx.x < off) red[threadIdx.x] += red[threadIdx.x + off];
        __syncthreads();
    }
    if (threadIdx.x == 0) {
        sq[row] = red[0];
        negmin[row] = 0x7F800000u;   // +inf
    }
}

// ---------------------------------------------------------------------------
// Kernel 2a (fast path): Gram tile from pre-split bf16 planes.
// TDM double-buffered staging: DMA of chunk k+1 overlaps WMMA of chunk k.
// ---------------------------------------------------------------------------
__global__ __launch_bounds__(256)
void gram_negmin_pre(const unsigned short* __restrict__ xhi,
                     const unsigned short* __restrict__ xlo,
                     const float* __restrict__ sq, const int* __restrict__ tgt,
                     unsigned* __restrict__ negmin) {
    __shared__ __align__(16) unsigned short As_hi[2][BM * LDK];
    __shared__ __align__(16) unsigned short As_lo[2][BM * LDK];
    __shared__ __align__(16) unsigned short Bs_hi[2][BN * LDK];
    __shared__ __align__(16) unsigned short Bs_lo[2][BN * LDK];

    const int bm   = blockIdx.x * BM;
    const int bn   = blockIdx.y * BN;
    const int tid  = threadIdx.x;
    const int wave = tid >> 5;
    const int lane = tid & 31;
    const int wm   = (wave >> 1) * 32;
    const int wn   = (wave & 1) * 32;
    const int lrow = lane & 15;
    const int kb   = (lane < 16) ? 0 : 8;

    v8f acc[2][2] = {};

#if HAVE_TDM
    const unsigned short* gahi = xhi + (size_t)bm * DIM;
    const unsigned short* galo = xlo + (size_t)bm * DIM;
    const unsigned short* gbhi = xhi + (size_t)bn * DIM;
    const unsigned short* gblo = xlo + (size_t)bn * DIM;

    if (wave == 0) {   // stage chunk 0 into buffer 0
        tdm_load_2d((unsigned)(uintptr_t)&As_hi[0][0], gahi, BM);
        tdm_load_2d((unsigned)(uintptr_t)&As_lo[0][0], galo, BM);
        tdm_load_2d((unsigned)(uintptr_t)&Bs_hi[0][0], gbhi, BN);
        tdm_load_2d((unsigned)(uintptr_t)&Bs_lo[0][0], gblo, BN);
        __builtin_amdgcn_s_wait_tensorcnt(0);
    }
    __syncthreads();

    for (int step = 0; step < KSTEPS; ++step) {
        int cur = step & 1;
        if (wave == 0 && step + 1 < KSTEPS) {   // DMA next chunk into other buffer
            int k1 = (step + 1) * BK;
            tdm_load_2d((unsigned)(uintptr_t)&As_hi[cur ^ 1][0], gahi + k1, BM);
            tdm_load_2d((unsigned)(uintptr_t)&As_lo[cur ^ 1][0], galo + k1, BM);
            tdm_load_2d((unsigned)(uintptr_t)&Bs_hi[cur ^ 1][0], gbhi + k1, BN);
            tdm_load_2d((unsigned)(uintptr_t)&Bs_lo[cur ^ 1][0], gblo + k1, BN);
        }
        wave_mma_step(As_hi[cur], As_lo[cur], Bs_hi[cur], Bs_lo[cur],
                      wm, wn, lrow, kb, acc);
        if (wave == 0)
            __builtin_amdgcn_s_wait_tensorcnt(0);
        __syncthreads();
    }
#else
    // Register-staged double-buffer fallback (no TDM builtin available)
    for (int step = 0; step < KSTEPS; ++step) {
        int cur = step & 1;
        int k0  = step * BK;
        #pragma unroll
        for (int it = 0; it < 2; ++it) {
            int unit = tid + it * 256;
            int r  = unit >> 2;
            int kc = (unit & 3) * 8;
            size_t goff = (size_t)(bm + r) * DIM + k0 + kc;
            *(v8us*)&As_hi[cur][r * LDK + kc] = *(const v8us*)(xhi + goff);
            *(v8us*)&As_lo[cur][r * LDK + kc] = *(const v8us*)(xlo + goff);
        }
        {
            int r  = tid >> 2;
            int kc = (tid & 3) * 8;
            size_t goff = (size_t)(bn + r) * DIM + k0 + kc;
            *(v8us*)&Bs_hi[cur][r * LDK + kc] = *(const v8us*)(xhi + goff);
            *(v8us*)&Bs_lo[cur][r * LDK + kc] = *(const v8us*)(xlo + goff);
        }
        __syncthreads();
        wave_mma_step(As_hi[cur], As_lo[cur], Bs_hi[cur], Bs_lo[cur],
                      wm, wn, lrow, kb, acc);
        __syncthreads();
    }
#endif
    wave_epilogue(acc, sq, tgt, negmin, bm, bn, wm, wn, lane, lrow);
}

// ---------------------------------------------------------------------------
// Kernel 2b (fallback if ws too small): convert fp32 -> bf16 hi/lo in-kernel
// ---------------------------------------------------------------------------
__global__ __launch_bounds__(256)
void gram_negmin_cvt(const float* __restrict__ x, const float* __restrict__ sq,
                     const int* __restrict__ tgt, unsigned* __restrict__ negmin) {
    __shared__ __align__(16) unsigned short As_hi[BM * LDK];
    __shared__ __align__(16) unsigned short As_lo[BM * LDK];
    __shared__ __align__(16) unsigned short Bs_hi[BN * LDK];
    __shared__ __align__(16) unsigned short Bs_lo[BN * LDK];

    const int bm   = blockIdx.x * BM;
    const int bn   = blockIdx.y * BN;
    const int tid  = threadIdx.x;
    const int wave = tid >> 5;
    const int lane = tid & 31;
    const int wm   = (wave >> 1) * 32;
    const int wn   = (wave & 1) * 32;
    const int lrow = lane & 15;
    const int kb   = (lane < 16) ? 0 : 8;

    v8f acc[2][2] = {};

    for (int k0 = 0; k0 < DIM; k0 += BK) {
        #pragma unroll
        for (int it = 0; it < 4; ++it) {
            int unit = tid + it * 256;
            int r  = unit >> 3;
            int kq = (unit & 7) * 4;
            const float* src = x + (size_t)(bm + r) * DIM + k0 + kq;
            v4f v = *(const v4f*)src;
            if (k0 + BK < DIM) __builtin_prefetch(src + BK, 0, 3);
            #pragma unroll
            for (int e = 0; e < 4; ++e) {
                unsigned short h = f2bf(v[e]);
                As_hi[r * LDK + kq + e] = h;
                As_lo[r * LDK + kq + e] = f2bf(v[e] - bf2f(h));
            }
        }
        #pragma unroll
        for (int it = 0; it < 2; ++it) {
            int unit = tid + it * 256;
            int r  = unit >> 3;
            int kq = (unit & 7) * 4;
            const float* src = x + (size_t)(bn + r) * DIM + k0 + kq;
            v4f v = *(const v4f*)src;
            if (k0 + BK < DIM) __builtin_prefetch(src + BK, 0, 3);
            #pragma unroll
            for (int e = 0; e < 4; ++e) {
                unsigned short h = f2bf(v[e]);
                Bs_hi[r * LDK + kq + e] = h;
                Bs_lo[r * LDK + kq + e] = f2bf(v[e] - bf2f(h));
            }
        }
        __syncthreads();
        wave_mma_step(As_hi, As_lo, Bs_hi, Bs_lo, wm, wn, lrow, kb, acc);
        __syncthreads();
    }
    wave_epilogue(acc, sq, tgt, negmin, bm, bn, wm, wn, lane, lrow);
}

// ---------------------------------------------------------------------------
// Kernel 3: within-identity anchor-positive distances (one wave per pair)
// ---------------------------------------------------------------------------
__global__ __launch_bounds__(256)
void within_pairs(const float* __restrict__ x, const float* __restrict__ sq,
                  float* __restrict__ dist_ap) {
    const int jj[NPAIR] = {0, 0, 0, 1, 1, 2};
    const int pp[NPAIR] = {1, 2, 3, 2, 3, 3};
    int pair = blockIdx.x * 8 + (threadIdx.x >> 5);
    int lane = threadIdx.x & 31;
    int p  = pair / NPAIR;
    int pr = pair % NPAIR;
    int a  = p * KID + jj[pr];
    int b  = p * KID + pp[pr];
    const float* xa = x + (size_t)a * DIM;
    const float* xb = x + (size_t)b * DIM;
    float s = 0.f;
    for (int k = lane * 4; k < DIM; k += 32 * 4) {
        v4f va = *(const v4f*)(xa + k);
        v4f vb = *(const v4f*)(xb + k);
        s += va.x * vb.x + va.y * vb.y + va.z * vb.z + va.w * vb.w;
    }
    #pragma unroll
    for (int off = 16; off >= 1; off >>= 1) s += __shfl_xor(s, off, 32);
    if (lane == 0) {
        float d2 = sq[a] + sq[b] - 2.0f * s;
        dist_ap[pair] = sqrtf(fmaxf(d2, 1e-12f));
    }
}

// ---------------------------------------------------------------------------
// Kernel 4: MarginRankingLoss mean + precision
// ---------------------------------------------------------------------------
__global__ __launch_bounds__(256)
void finalize(const float* __restrict__ dist_ap,
              const unsigned* __restrict__ negmin,
              float* __restrict__ out) {
    const int jj[NPAIR] = {0, 0, 0, 1, 1, 2};
    float lsum = 0.f, psum = 0.f;
    for (int i = threadIdx.x; i < NP_TOT; i += 256) {
        int p  = i / NPAIR;
        int pr = i % NPAIR;
        float ap = dist_ap[i];
        float an = __uint_as_float(negmin[p * KID + jj[pr]]);
        lsum += fmaxf(ap - an + MARGIN, 0.f);
        psum += (an > ap) ? 1.f : 0.f;
    }
    __shared__ float rl[256], rp[256];
    rl[threadIdx.x] = lsum;
    rp[threadIdx.x] = psum;
    __syncthreads();
    for (int off = 128; off > 0; off >>= 1) {
        if (threadIdx.x < off) {
            rl[threadIdx.x] += rl[threadIdx.x + off];
            rp[threadIdx.x] += rp[threadIdx.x + off];
        }
        __syncthreads();
    }
    if (threadIdx.x == 0) {
        out[0] = rl[0] / (float)NP_TOT;
        out[1] = rp[0] / (float)NP_TOT;
    }
}

// ---------------------------------------------------------------------------
extern "C" void kernel_launch(void* const* d_in, const int* in_sizes, int n_in,
                              void* d_out, int out_size, void* d_ws, size_t ws_size,
                              hipStream_t stream) {
    const float* x   = (const float*)d_in[0];
    const int*   tgt = (const int*)d_in[1];
    // d_in[2] = epoch (unused by the loss)
    float*    out    = (float*)d_out;
    float*    sq     = (float*)((char*)d_ws + WS_SQ);
    unsigned* negmin = (unsigned*)((char*)d_ws + WS_NEGMIN);
    float*    ap     = (float*)((char*)d_ws + WS_AP);

    rownorm_init<<<N_ROWS, 256, 0, stream>>>(x, sq, negmin);

    dim3 grid(N_ROWS / BM, N_ROWS / BN);
    if (ws_size >= WS_NEED) {
        unsigned short* xhi = (unsigned short*)((char*)d_ws + WS_HI);
        unsigned short* xlo = (unsigned short*)((char*)d_ws + WS_LO);
        xsplit<<<(N_ROWS * DIM) / (256 * 8), 256, 0, stream>>>(x, xhi, xlo);
        gram_negmin_pre<<<grid, 256, 0, stream>>>(xhi, xlo, sq, tgt, negmin);
    } else {
        gram_negmin_cvt<<<grid, 256, 0, stream>>>(x, sq, tgt, negmin);
    }

    within_pairs<<<NP_TOT / 8, 256, 0, stream>>>(x, sq, ap);

    finalize<<<1, 256, 0, stream>>>(ap, negmin, out);
}
